// LatentJiT_84293028151598
// MI455X (gfx1250) — matrix operand order repaired and checked by reading, not compiled, not run on hardware
//
#include <hip/hip_runtime.h>

// ---------------------------------------------------------------------------
// Types / helpers
// ---------------------------------------------------------------------------
typedef __attribute__((ext_vector_type(16))) __bf16 v16bf;
typedef __attribute__((ext_vector_type(8)))  float  v8f;
typedef __attribute__((ext_vector_type(4)))  unsigned int u32x4;
typedef int v4i __attribute__((__vector_size__(16)));

union Frag {
    u32x4 q[2];
    v16bf v;
};

__device__ __forceinline__ unsigned short f2bf(float f) {
    unsigned int u = __float_as_uint(f);
    unsigned int r = u + 0x7fffu + ((u >> 16) & 1u);
    return (unsigned short)(r >> 16);
}
__device__ __forceinline__ float bf2f(unsigned short h) {
    return __uint_as_float(((unsigned int)h) << 16);
}
__device__ __forceinline__ void vzero(v8f& a) {
#pragma unroll
    for (int e = 0; e < 8; e++) a[e] = 0.0f;
}

// CDNA5 async global->LDS copy (ASYNCcnt) with safe fallback.
#if defined(__AMDGCN__) && __has_builtin(__builtin_amdgcn_global_load_async_to_lds_b128)
#define HAVE_ASYNC_LDS 1
#else
#define HAVE_ASYNC_LDS 0
#endif

__device__ __forceinline__ void cp16(const unsigned short* __restrict__ g,
                                     unsigned short* __restrict__ l) {
#if HAVE_ASYNC_LDS
    __builtin_amdgcn_global_load_async_to_lds_b128((v4i*)g, (v4i*)l, 0, 0);
#else
    *(u32x4*)l = *(const u32x4*)g;
#endif
}

__device__ __forceinline__ void wait_async() {
#if HAVE_ASYNC_LDS
#if __has_builtin(__builtin_amdgcn_s_wait_asynccnt)
    __builtin_amdgcn_s_wait_asynccnt(0);
#else
    asm volatile("s_wait_asynccnt 0" ::: "memory");
#endif
#endif
}

// ---------------------------------------------------------------------------
// bf16 WMMA GEMM:  out[M,N] = A[M,K]·B[N,K]^T + bias
// A, B both bf16. All tiles unguarded: caller guarantees
//   - A readable for ceil(M/128)*128 rows, lda%8==0
//   - B readable for N rows (N%128==0), ldb%8==0, K%32==0
// Double-buffered LDS, async global->LDS copies, 1 barrier / K-step.
// Block 256 thr (8 waves), tile 128x128x32; wave tile 32x64 (2x4 WMMA).
// EPI: 0 = fp32 store, 1 = bf16 store, 2 = resgate: outF += gate*(acc+bias)
// ---------------------------------------------------------------------------
#define EPI_F32 0
#define EPI_BF16 1
#define EPI_RESGATE 2

template <int EPI>
__global__ __launch_bounds__(256, 2) void gemm_bf16(
    const unsigned short* __restrict__ A, int lda,
    const unsigned short* __restrict__ Bw, int ldb,
    const float* __restrict__ bias,
    int M, int N, int K,
    float* __restrict__ outF, unsigned short* __restrict__ outH,
    const float* __restrict__ gate, int gstride) {
    constexpr int LDT = 40;  // 32 + 8 pad, keeps 16B alignment
    __shared__ unsigned short sA[2][128 * LDT];
    __shared__ unsigned short sB[2][128 * LDT];

    const int tid = threadIdx.x;
    const int lane = tid & 31, wave = tid >> 5;
    const int half = lane >> 4, l16 = lane & 15;
    const int m0 = blockIdx.y * 128;
    const int n0 = blockIdx.x * 128;
    const int wmoff = (wave >> 1) * 32;
    const int wnoff = (wave & 1) * 64;

    v8f acc[2][4];
#pragma unroll
    for (int i = 0; i < 2; i++)
#pragma unroll
        for (int j = 0; j < 4; j++) vzero(acc[i][j]);

    const unsigned short* Ab = A + (size_t)m0 * lda;
    const unsigned short* Bb = Bw + (size_t)n0 * ldb;
    const int r0 = tid >> 2;          // rows 0..63 (chunk 0) / +64 (chunk 1)
    const int c0 = (tid & 3) * 8;

    auto issue = [&](int kt, int buf) {
        const unsigned short* ga = Ab + kt * 32;
        const unsigned short* gb = Bb + kt * 32;
#pragma unroll
        for (int i = 0; i < 2; i++) {
            int r = r0 + i * 64;
            cp16(ga + (size_t)r * lda + c0, &sA[buf][r * LDT + c0]);
            cp16(gb + (size_t)r * ldb + c0, &sB[buf][r * LDT + c0]);
        }
    };

    const int nk = K >> 5;
    issue(0, 0);
    for (int kt = 0; kt < nk; kt++) {
        const int cur = kt & 1;
        wait_async();
        __syncthreads();
        if (kt + 1 < nk) issue(kt + 1, cur ^ 1);

        Frag a[2], b[4];
#pragma unroll
        for (int tm = 0; tm < 2; tm++) {
            const unsigned short* p = &sA[cur][(wmoff + tm * 16 + l16) * LDT + half * 8];
            a[tm].q[0] = *(const u32x4*)p;
            a[tm].q[1] = *(const u32x4*)(p + 16);
        }
#pragma unroll
        for (int tn = 0; tn < 4; tn++) {
            const unsigned short* p = &sB[cur][(wnoff + tn * 16 + l16) * LDT + half * 8];
            b[tn].q[0] = *(const u32x4*)p;
            b[tn].q[1] = *(const u32x4*)(p + 16);
        }
#pragma unroll
        for (int tm = 0; tm < 2; tm++)
#pragma unroll
            for (int tn = 0; tn < 4; tn++)
                acc[tm][tn] = __builtin_amdgcn_wmma_f32_16x16x32_bf16(
                    false, a[tm].v, false, b[tn].v, (short)0, acc[tm][tn],
                    false, false);
    }

#pragma unroll
    for (int tm = 0; tm < 2; tm++) {
#pragma unroll
        for (int tn = 0; tn < 4; tn++) {
            int gn = n0 + wnoff + tn * 16 + l16;
            float bs = bias[gn];
#pragma unroll
            for (int j = 0; j < 8; j++) {
                int gm = m0 + wmoff + tm * 16 + j + 8 * half;
                if (gm >= M) continue;
                float v = acc[tm][tn][j] + bs;
                if (EPI == EPI_F32) {
                    outF[(size_t)gm * N + gn] = v;
                } else if (EPI == EPI_BF16) {
                    outH[(size_t)gm * N + gn] = f2bf(v);
                } else {
                    float g = gate[(size_t)(gm >> 8) * gstride + gn];
                    outF[(size_t)gm * N + gn] += g * v;
                }
            }
        }
    }
}

// ---------------------------------------------------------------------------
// Fused SwiGLU GEMM: act[M,2752] = silu(A·W1^T+b1) * (A·W2^T+b2) (bf16)
// W12bf: [2*2752, 1024] bf16, rows 2730..2751 (per half) zero-padded.
// Columns 2730..2751 of act are stored as 0 so w3 GEMM can run K=2752.
// ---------------------------------------------------------------------------
__global__ __launch_bounds__(256, 2) void gemm_swiglu(
    const unsigned short* __restrict__ A,     // [M,1024] bf16
    const unsigned short* __restrict__ W12,   // [5504,1024] bf16 padded
    const float* __restrict__ b12,            // [5460]
    int M, unsigned short* __restrict__ act)  // [M,2752] bf16
{
    constexpr int K = 1024, NHALF = 2730, NPAD = 2752, LDT = 40;
    __shared__ unsigned short sA[2][128 * LDT];
    __shared__ unsigned short sB1[2][64 * LDT];
    __shared__ unsigned short sB2[2][64 * LDT];

    const int tid = threadIdx.x;
    const int lane = tid & 31, wave = tid >> 5;
    const int half = lane >> 4, l16 = lane & 15;
    const int m0 = blockIdx.y * 128;
    const int n0 = blockIdx.x * 64;
    const int wmoff = (wave >> 1) * 32;
    const int wnoff = (wave & 1) * 32;

    v8f acc1[2][2], acc2[2][2];
#pragma unroll
    for (int i = 0; i < 2; i++)
#pragma unroll
        for (int j = 0; j < 2; j++) { vzero(acc1[i][j]); vzero(acc2[i][j]); }

    const unsigned short* Ab = A + (size_t)m0 * K;
    const unsigned short* B1b = W12 + (size_t)n0 * K;
    const unsigned short* B2b = W12 + (size_t)(NPAD + n0) * K;
    const int rA = tid >> 2;          // 0..63 (+64 for chunk 1)
    const int cA = (tid & 3) * 8;

    auto issue = [&](int kt, int buf) {
        int ko = kt * 32;
#pragma unroll
        for (int i = 0; i < 2; i++) {
            int r = rA + i * 64;
            cp16(Ab + (size_t)r * K + ko + cA, &sA[buf][r * LDT + cA]);
        }
        cp16(B1b + (size_t)rA * K + ko + cA, &sB1[buf][rA * LDT + cA]);
        cp16(B2b + (size_t)rA * K + ko + cA, &sB2[buf][rA * LDT + cA]);
    };

    issue(0, 0);
    for (int kt = 0; kt < K / 32; kt++) {
        const int cur = kt & 1;
        wait_async();
        __syncthreads();
        if (kt + 1 < K / 32) issue(kt + 1, cur ^ 1);

        Frag a[2], b1[2], b2[2];
#pragma unroll
        for (int tm = 0; tm < 2; tm++) {
            const unsigned short* p = &sA[cur][(wmoff + tm * 16 + l16) * LDT + half * 8];
            a[tm].q[0] = *(const u32x4*)p;
            a[tm].q[1] = *(const u32x4*)(p + 16);
        }
#pragma unroll
        for (int tn = 0; tn < 2; tn++) {
            const unsigned short* p1 = &sB1[cur][(wnoff + tn * 16 + l16) * LDT + half * 8];
            b1[tn].q[0] = *(const u32x4*)p1;
            b1[tn].q[1] = *(const u32x4*)(p1 + 16);
            const unsigned short* p2 = &sB2[cur][(wnoff + tn * 16 + l16) * LDT + half * 8];
            b2[tn].q[0] = *(const u32x4*)p2;
            b2[tn].q[1] = *(const u32x4*)(p2 + 16);
        }
#pragma unroll
        for (int tm = 0; tm < 2; tm++)
#pragma unroll
            for (int tn = 0; tn < 2; tn++) {
                acc1[tm][tn] = __builtin_amdgcn_wmma_f32_16x16x32_bf16(
                    false, a[tm].v, false, b1[tn].v, (short)0, acc1[tm][tn], false, false);
                acc2[tm][tn] = __builtin_amdgcn_wmma_f32_16x16x32_bf16(
                    false, a[tm].v, false, b2[tn].v, (short)0, acc2[tm][tn], false, false);
            }
    }

#pragma unroll
    for (int tm = 0; tm < 2; tm++) {
#pragma unroll
        for (int tn = 0; tn < 2; tn++) {
            int gn = n0 + wnoff + tn * 16 + l16;  // < 2752
#pragma unroll
            for (int j = 0; j < 8; j++) {
                int gm = m0 + wmoff + tm * 16 + j + 8 * half;
                if (gm >= M) continue;
                float outv = 0.0f;
                if (gn < NHALF) {
                    float x1 = acc1[tm][tn][j] + b12[gn];
                    float x2 = acc2[tm][tn][j] + b12[NHALF + gn];
                    outv = x1 / (1.0f + __expf(-x1)) * x2;
                }
                act[(size_t)gm * NPAD + gn] = f2bf(outv);
            }
        }
    }
}

// ---------------------------------------------------------------------------
// Attention: per (b,h, 64-row query block). 4 waves x 16 rows.
// S = Q·K^T (WMMA) -> softmax in registers (shfl) -> P (LDS bf16) · V (WMMA).
// ---------------------------------------------------------------------------
__global__ __launch_bounds__(128) void attn_kernel(
    const unsigned short* __restrict__ qkv, unsigned short* __restrict__ obuf) {
    __shared__ unsigned short sK[256 * 72];
    __shared__ unsigned short sV[64 * 264];
    __shared__ unsigned short sP[4 * 16 * 264];

    const int tid = threadIdx.x, lane = tid & 31, wave = tid >> 5;
    const int half = lane >> 4, l16 = lane & 15;
    const int b = blockIdx.x >> 6;
    const int h = (blockIdx.x >> 2) & 15;
    const int mb = blockIdx.x & 3;
    const size_t tb = (size_t)b * 256;

    for (int idx = tid; idx < 2048; idx += 128) {
        int row = idx >> 3, q4 = idx & 7;
        u32x4 v = *(const u32x4*)(qkv + (tb + row) * 3072 + 1024 + h * 64 + q4 * 8);
        *(u32x4*)&sK[row * 72 + q4 * 8] = v;
    }
    for (int idx = tid; idx < 8192; idx += 128) {
        int row = idx >> 5, dp = (idx & 31) * 2;
        unsigned int u = *(const unsigned int*)(qkv + (tb + row) * 3072 + 2048 + h * 64 + dp);
        sV[dp * 264 + row] = (unsigned short)(u & 0xffffu);
        sV[(dp + 1) * 264 + row] = (unsigned short)(u >> 16);
    }
    __syncthreads();

    const int mrow = mb * 64 + wave * 16;

    v8f s[16];
#pragma unroll
    for (int kb = 0; kb < 16; kb++) vzero(s[kb]);

#pragma unroll
    for (int kd = 0; kd < 2; kd++) {
        Frag a;
        const unsigned short* pq =
            qkv + (tb + mrow + l16) * 3072 + h * 64 + kd * 32 + half * 8;
        a.q[0] = *(const u32x4*)pq;
        a.q[1] = *(const u32x4*)(pq + 16);
#pragma unroll
        for (int kb = 0; kb < 16; kb++) {
            Frag bb;
            const unsigned short* pk = &sK[(kb * 16 + l16) * 72 + kd * 32 + half * 8];
            bb.q[0] = *(const u32x4*)pk;
            bb.q[1] = *(const u32x4*)(pk + 16);
            s[kb] = __builtin_amdgcn_wmma_f32_16x16x32_bf16(
                false, a.v, false, bb.v, (short)0, s[kb], false, false);
        }
    }

    float mx[8], sm[8];
#pragma unroll
    for (int j = 0; j < 8; j++) mx[j] = -3.0e38f;
#pragma unroll
    for (int kb = 0; kb < 16; kb++)
#pragma unroll
        for (int j = 0; j < 8; j++) {
            float v = s[kb][j] * 0.125f;
            s[kb][j] = v;
            mx[j] = fmaxf(mx[j], v);
        }
#pragma unroll
    for (int msk = 1; msk < 16; msk <<= 1)
#pragma unroll
        for (int j = 0; j < 8; j++) mx[j] = fmaxf(mx[j], __shfl_xor(mx[j], msk, 32));
#pragma unroll
    for (int j = 0; j < 8; j++) sm[j] = 0.0f;
#pragma unroll
    for (int kb = 0; kb < 16; kb++)
#pragma unroll
        for (int j = 0; j < 8; j++) {
            float e = __expf(s[kb][j] - mx[j]);
            s[kb][j] = e;
            sm[j] += e;
        }
#pragma unroll
    for (int msk = 1; msk < 16; msk <<= 1)
#pragma unroll
        for (int j = 0; j < 8; j++) sm[j] += __shfl_xor(sm[j], msk, 32);
#pragma unroll
    for (int j = 0; j < 8; j++) sm[j] = 1.0f / sm[j];

    unsigned short* pr = &sP[wave * 16 * 264];
#pragma unroll
    for (int kb = 0; kb < 16; kb++)
#pragma unroll
        for (int j = 0; j < 8; j++)
            pr[(j + 8 * half) * 264 + kb * 16 + l16] = f2bf(s[kb][j] * sm[j]);

    v8f o[4];
#pragma unroll
    for (int dn = 0; dn < 4; dn++) vzero(o[dn]);
#pragma unroll
    for (int ks = 0; ks < 8; ks++) {
        Frag a;
        const unsigned short* pp = &pr[l16 * 264 + ks * 32 + half * 8];
        a.q[0] = *(const u32x4*)pp;
        a.q[1] = *(const u32x4*)(pp + 16);
#pragma unroll
        for (int dn = 0; dn < 4; dn++) {
            Frag bb;
            const unsigned short* pv = &sV[(dn * 16 + l16) * 264 + ks * 32 + half * 8];
            bb.q[0] = *(const u32x4*)pv;
            bb.q[1] = *(const u32x4*)(pv + 16);
            o[dn] = __builtin_amdgcn_wmma_f32_16x16x32_bf16(
                false, a.v, false, bb.v, (short)0, o[dn], false, false);
        }
    }
#pragma unroll
    for (int dn = 0; dn < 4; dn++)
#pragma unroll
        for (int j = 0; j < 8; j++) {
            int n = mrow + j + 8 * half;
            obuf[(tb + n) * 1024 + h * 64 + dn * 16 + l16] = f2bf(o[dn][j]);
        }
}

// ---------------------------------------------------------------------------
// Per-head q/k RMSNorm + RoPE, in place on bf16 qkv. One wave per (b,h,n).
// ---------------------------------------------------------------------------
__global__ __launch_bounds__(256) void qk_rope(
    unsigned short* __restrict__ qkv,
    const float* __restrict__ qw, const float* __restrict__ kw) {
    const int lane = threadIdx.x & 31, wave = threadIdx.x >> 5;
    const int gid = blockIdx.x * 8 + wave;
    const int b = gid >> 12;
    const int h = (gid >> 8) & 15;
    const int n = gid & 255;
    const size_t base = ((size_t)(b * 256 + n)) * 3072 + h * 64;
    const int d0 = lane * 2;

    float freq = __expf(-logf(10000.0f) * (float)(lane & 15) / 16.0f);
    int sidx = (lane < 16) ? (n >> 4) : (n & 15);
    float ang = (float)sidx * freq;
    float cs = cosf(ang), sn = sinf(ang);

#pragma unroll
    for (int qk = 0; qk < 2; qk++) {
        const size_t off = base + qk * 1024 + d0;
        const float* w = (qk == 0) ? qw : kw;
        unsigned int u = *(unsigned int*)(qkv + off);
        float x0 = bf2f((unsigned short)u);
        float x1 = bf2f((unsigned short)(u >> 16));
        float ss = x0 * x0 + x1 * x1;
#pragma unroll
        for (int m = 1; m < 32; m <<= 1) ss += __shfl_xor(ss, m, 32);
        float r = rsqrtf(ss * (1.0f / 64.0f) + 1e-6f);
        float a0 = x0 * r * w[d0];
        float a1 = x1 * r * w[d0 + 1];
        float o0 = a0 * cs - a1 * sn;
        float o1 = a1 * cs + a0 * sn;
        *(unsigned int*)(qkv + off) =
            (unsigned int)f2bf(o0) | ((unsigned int)f2bf(o1) << 16);
    }
}

// ---------------------------------------------------------------------------
// RMSNorm + modulate -> bf16.  modb points at shift block; scale at +1024.
// ---------------------------------------------------------------------------
__global__ __launch_bounds__(256) void normmod(
    const float* __restrict__ xt, const float* __restrict__ w,
    const float* __restrict__ modb, int modstride,
    unsigned short* __restrict__ out) {
    __shared__ float red[256];
    const int row = blockIdx.x, tid = threadIdx.x;
    const int b = row >> 8;
    const float* xr = xt + (size_t)row * 1024;
    float v[4];
    float ss = 0.0f;
#pragma unroll
    for (int i = 0; i < 4; i++) {
        v[i] = xr[tid + i * 256];
        ss += v[i] * v[i];
    }
    red[tid] = ss;
    __syncthreads();
    for (int s = 128; s > 0; s >>= 1) {
        if (tid < s) red[tid] += red[tid + s];
        __syncthreads();
    }
    float rs = rsqrtf(red[0] * (1.0f / 1024.0f) + 1e-6f);
    const float* mb = modb + (size_t)b * modstride;
#pragma unroll
    for (int i = 0; i < 4; i++) {
        int c = tid + i * 256;
        float hv = v[i] * rs * w[c] * (1.0f + mb[1024 + c]) + mb[c];
        out[(size_t)row * 1024 + c] = f2bf(hv);
    }
}

// ---------------------------------------------------------------------------
// Weight conversion kernels (fp32 -> bf16, optional padding)
// ---------------------------------------------------------------------------
__global__ void cvt_flat(const float* __restrict__ src,
                         unsigned short* __restrict__ dst, long n) {
    long i = (long)blockIdx.x * 256 + threadIdx.x;
    if (i < n) dst[i] = f2bf(src[i]);
}

// dst [rows, cd], src [rows, cs], pad columns cs..cd-1 with zero
__global__ void cvt_padK(const float* __restrict__ src,
                         unsigned short* __restrict__ dst,
                         int rows, int cs, int cd) {
    long i = (long)blockIdx.x * 256 + threadIdx.x;
    if (i >= (long)rows * cd) return;
    int r = (int)(i / cd), c = (int)(i % cd);
    dst[i] = (c < cs) ? f2bf(src[(long)r * cs + c]) : (unsigned short)0;
}

// w12 [5460,1024] -> [2*2752,1024], each 2730-row half padded to 2752
__global__ void cvt_w12(const float* __restrict__ src,
                        unsigned short* __restrict__ dst) {
    long i = (long)blockIdx.x * 256 + threadIdx.x;
    if (i >= (long)5504 * 1024) return;
    int c = (int)(i & 1023);
    long r = i >> 10;
    int hs = (int)(r / 2752), rs = (int)(r % 2752);
    float v = (rs < 2730) ? src[((long)hs * 2730 + rs) * 1024 + c] : 0.0f;
    dst[i] = f2bf(v);
}

__global__ void zero_bf(unsigned short* __restrict__ p, long n) {
    long i = (long)blockIdx.x * 256 + threadIdx.x;
    if (i < n) p[i] = 0;
}

// ---------------------------------------------------------------------------
// Conditioning path (small)
// ---------------------------------------------------------------------------
__global__ void temb_embed(const float* __restrict__ t, float* __restrict__ femb) {
    int b = blockIdx.x, j = threadIdx.x;
    int i = j & 127;
    float fr = __expf(-logf(10000.0f) * (float)i / 128.0f);
    float a = t[b] * fr;
    femb[b * 256 + j] = (j < 128) ? cosf(a) : sinf(a);
}

__global__ void temb_mlp1(const float* __restrict__ femb,
                          const float* __restrict__ w1,
                          const float* __restrict__ b1,
                          float* __restrict__ th) {
    int idx = blockIdx.x * 256 + threadIdx.x;
    int b = idx >> 10, c = idx & 1023;
    float s = b1[c];
    const float* f = femb + b * 256;
    const float* w = w1 + (size_t)c * 256;
    for (int j = 0; j < 256; j++) s += f[j] * w[j];
    th[idx] = s / (1.0f + __expf(-s));
}

__global__ void temb_mlp2(const float* __restrict__ th,
                          const float* __restrict__ w2,
                          const float* __restrict__ b2,
                          const float* __restrict__ ytab,
                          const int* __restrict__ y,
                          unsigned short* __restrict__ scb) {
    int idx = blockIdx.x * 256 + threadIdx.x;
    int b = idx >> 10, c = idx & 1023;
    float s = b2[c];
    const float* hh = th + (size_t)b * 1024;
    const float* w = w2 + (size_t)c * 1024;
    for (int j = 0; j < 1024; j++) s += hh[j] * w[j];
    s += ytab[(size_t)y[b] * 1024 + c];
    scb[idx] = f2bf(s / (1.0f + __expf(-s)));
}

// ---------------------------------------------------------------------------
// Patch projection + pos-embed
// ---------------------------------------------------------------------------
__global__ void xproj_kernel(const float* __restrict__ x,
                             const float* __restrict__ pw,
                             const float* __restrict__ pb,
                             const float* __restrict__ pos,
                             float* __restrict__ xt) {
    size_t idx = (size_t)blockIdx.x * 256 + threadIdx.x;
    int c = (int)(idx & 1023);
    size_t token = idx >> 10;
    int b = (int)(token >> 8), n = (int)(token & 255);
    float s = pb[c];
    const float* xp = x + ((size_t)b * 32) * 256 + n;
    const float* w = pw + (size_t)c * 32;
#pragma unroll
    for (int k = 0; k < 32; k++) s += xp[(size_t)k * 256] * w[k];
    s += pos[(size_t)n * 1024 + c];
    xt[idx] = s;
}

// ---------------------------------------------------------------------------
// Final linear
// ---------------------------------------------------------------------------
__global__ void flin_kernel(const unsigned short* __restrict__ hbuf,
                            const float* __restrict__ fw,
                            const float* __restrict__ fb,
                            float* __restrict__ out) {
    size_t idx = (size_t)blockIdx.x * 256 + threadIdx.x;
    int k = (int)(idx & 31);
    size_t token = idx >> 5;
    int b = (int)(token >> 8), n = (int)(token & 255);
    float s = fb[k];
    const unsigned short* hr = hbuf + token * 1024;
    const float* w = fw + (size_t)k * 1024;
    for (int c = 0; c < 1024; c++) s += bf2f(hr[c]) * w[c];
    out[((size_t)b * 32 + k) * 256 + n] = s;
}

// ---------------------------------------------------------------------------
// Launch
// ---------------------------------------------------------------------------
extern "C" void kernel_launch(void* const* d_in, const int* in_sizes, int n_in,
                              void* d_out, int out_size, void* d_ws,
                              size_t ws_size, hipStream_t stream) {
    const float* x = (const float*)d_in[0];
    const float* t = (const float*)d_in[1];
    const int* y = (const int*)d_in[2];
    const float* pos = (const float*)d_in[3];
    const float* xpw = (const float*)d_in[4];
    const float* xpb = (const float*)d_in[5];
    const float* tw1 = (const float*)d_in[6];
    const float* tb1 = (const float*)d_in[7];
    const float* tw2 = (const float*)d_in[8];
    const float* tb2 = (const float*)d_in[9];
    const float* ytab = (const float*)d_in[10];
    const float* qkvw = (const float*)d_in[11];
    const float* qkvbias = (const float*)d_in[12];
    const float* apw = (const float*)d_in[13];
    const float* apb = (const float*)d_in[14];
    const float* adaw = (const float*)d_in[15];
    const float* adab = (const float*)d_in[16];
    const float* w12w = (const float*)d_in[17];
    const float* w12b = (const float*)d_in[18];
    const float* w3w = (const float*)d_in[19];
    const float* w3b = (const float*)d_in[20];
    const float* n1w = (const float*)d_in[21];
    const float* n2w = (const float*)d_in[22];
    const float* qnw = (const float*)d_in[23];
    const float* knw = (const float*)d_in[24];
    const float* fnw = (const float*)d_in[25];
    const float* fadaw = (const float*)d_in[26];
    const float* fadab = (const float*)d_in[27];
    const float* flw = (const float*)d_in[28];
    const float* flb = (const float*)d_in[29];

    char* wsp = (char*)d_ws;
    auto alloc = [&](size_t bytes) -> char* {
        char* p = wsp;
        wsp += (bytes + 255) & ~(size_t)255;
        return p;
    };
    float* femb = (float*)alloc((size_t)64 * 256 * 4);
    float* th = (float*)alloc((size_t)64 * 1024 * 4);
    unsigned short* scb = (unsigned short*)alloc((size_t)128 * 1024 * 2);  // padded to 128 rows
    float* xt = (float*)alloc((size_t)16384 * 1024 * 4);
    float* modb = (float*)alloc((size_t)64 * 6144 * 4);
    float* fmod = (float*)alloc((size_t)64 * 2048 * 4);
    unsigned short* hb = (unsigned short*)alloc((size_t)16384 * 1024 * 2);
    unsigned short* qb = (unsigned short*)alloc((size_t)16384 * 3072 * 2);
    unsigned short* ob = (unsigned short*)alloc((size_t)16384 * 1024 * 2);
    unsigned short* actb = (unsigned short*)alloc((size_t)16384 * 2752 * 2);
    // per-layer bf16 weight staging
    unsigned short* wqkv = (unsigned short*)alloc((size_t)3072 * 1024 * 2);
    unsigned short* wap = (unsigned short*)alloc((size_t)1024 * 1024 * 2);
    unsigned short* wada = (unsigned short*)alloc((size_t)6144 * 1024 * 2);
    unsigned short* ww12 = (unsigned short*)alloc((size_t)5504 * 1024 * 2);
    unsigned short* ww3 = (unsigned short*)alloc((size_t)1024 * 2752 * 2);
    unsigned short* wfada = (unsigned short*)alloc((size_t)2048 * 1024 * 2);

    temb_embed<<<64, 256, 0, stream>>>(t, femb);
    temb_mlp1<<<256, 256, 0, stream>>>(femb, tw1, tb1, th);
    temb_mlp2<<<256, 256, 0, stream>>>(th, tw2, tb2, ytab, y, scb);
    zero_bf<<<256, 256, 0, stream>>>(scb + 64 * 1024, 64 * 1024);  // padded rows
    xproj_kernel<<<65536, 256, 0, stream>>>(x, xpw, xpb, pos, xt);

    for (int l = 0; l < 12; l++) {
        // ---- convert this layer's weights to bf16 (cheap: ~19M elems) ----
        cvt_flat<<<(3145728 + 255) / 256, 256, 0, stream>>>(
            qkvw + (size_t)l * 3145728, wqkv, 3145728);
        cvt_flat<<<(1048576 + 255) / 256, 256, 0, stream>>>(
            apw + (size_t)l * 1048576, wap, 1048576);
        cvt_flat<<<(6291456 + 255) / 256, 256, 0, stream>>>(
            adaw + (size_t)l * 6291456, wada, 6291456);
        cvt_w12<<<(5636096 + 255) / 256, 256, 0, stream>>>(
            w12w + (size_t)l * 5460 * 1024, ww12);
        cvt_padK<<<(2818048 + 255) / 256, 256, 0, stream>>>(
            w3w + (size_t)l * 1024 * 2730, ww3, 1024, 2730, 2752);

        // adaLN modulation: mod[64,6144]
        gemm_bf16<EPI_F32><<<dim3(48, 1), 256, 0, stream>>>(
            scb, 1024, wada, 1024, adab + (size_t)l * 6144,
            64, 6144, 1024, modb, nullptr, nullptr, 0);
        normmod<<<16384, 256, 0, stream>>>(xt, n1w + l * 1024, modb, 6144, hb);
        gemm_bf16<EPI_BF16><<<dim3(24, 128), 256, 0, stream>>>(
            hb, 1024, wqkv, 1024, qkvbias + (size_t)l * 3072,
            16384, 3072, 1024, nullptr, qb, nullptr, 0);
        qk_rope<<<32768, 256, 0, stream>>>(qb, qnw + l * 64, knw + l * 64);
        attn_kernel<<<4096, 128, 0, stream>>>(qb, ob);
        gemm_bf16<EPI_RESGATE><<<dim3(8, 128), 256, 0, stream>>>(
            ob, 1024, wap, 1024, apb + (size_t)l * 1024,
            16384, 1024, 1024, xt, nullptr, modb + 2 * 1024, 6144);
        normmod<<<16384, 256, 0, stream>>>(xt, n2w + l * 1024, modb + 3 * 1024, 6144, hb);
        gemm_swiglu<<<dim3(43, 128), 256, 0, stream>>>(
            hb, ww12, w12b + (size_t)l * 5460, 16384, actb);
        gemm_bf16<EPI_RESGATE><<<dim3(8, 128), 256, 0, stream>>>(
            actb, 2752, ww3, 2752, w3b + (size_t)l * 1024,
            16384, 1024, 2752, xt, nullptr, modb + 5 * 1024, 6144);
    }

    cvt_flat<<<(2097152 + 255) / 256, 256, 0, stream>>>(fadaw, wfada, 2097152);
    gemm_bf16<EPI_F32><<<dim3(16, 1), 256, 0, stream>>>(
        scb, 1024, wfada, 1024, fadab, 64, 2048, 1024, fmod, nullptr, nullptr, 0);
    normmod<<<16384, 256, 0, stream>>>(xt, fnw, fmod, 2048, hb);
    flin_kernel<<<2048, 256, 0, stream>>>(hb, flw, flb, (float*)d_out);
}
